// MIPNetwork_66013647340219
// MI455X (gfx1250) — compile-verified, matrix operand order checked by reading, not run on hardware
//
#include <hip/hip_runtime.h>
#include <hip/hip_bf16.h>
#include <math.h>

// ---------------- problem constants (from reference) ----------------
#define FM          32
#define OUT_BITS    16
#define PASS_STEPS  3
#define N_EDGES_C   3200000
#define VAR_COUNT_C 100000
#define CONST_COUNT_C 50000
#define EPS_C       1e-5f

typedef __attribute__((ext_vector_type(16))) _Float16 v16h;
typedef __attribute__((ext_vector_type(8)))  float    v8f;

// K index for half-element h of a 16-bit WMMA A/B fragment (16x32, wave32).
// Lanes 0-15: halves 0..7 -> K=0..7, halves 8..15 -> K=16..23
// Lanes 16-31: halves 0..7 -> K=8..15, halves 8..15 -> K=24..31
__device__ __forceinline__ int kmap(int lane, int h) {
    int kb = (lane >> 4) << 3;
    return (h < 8) ? (kb + h) : (16 + kb + (h - 8));
}

// ---------------- init ----------------
__global__ void fill_f32(float* __restrict__ p, float v, int n) {
    int i = blockIdx.x * blockDim.x + threadIdx.x;
    if (i < n) p[i] = v;
}

// ---------------- pack f32 weight (K x N, row-major) into f16 WMMA B-fragments ----
// Layout: frag ft = c*NT + t (c = K-chunk of 32, t = N-tile of 16);
// element [ft][lane][h] at out[((ft*32)+lane)*16 + h]  ->  one wave reads a
// fragment as a contiguous v16h (32B) per lane.
__global__ void pack_wgt(const float* __restrict__ W, _Float16* __restrict__ out,
                         int K, int N) {
    int idx = blockIdx.x * blockDim.x + threadIdx.x;
    int NT = N >> 4;
    int total = (K >> 5) * NT * 32 * 16;
    if (idx >= total) return;
    int h    = idx & 15;
    int lane = (idx >> 4) & 31;
    int ft   = idx >> 9;
    int t = ft % NT, c = ft / NT;
    int k = c * 32 + kmap(lane, h);
    int n = t * 16 + (lane & 15);
    out[idx] = (_Float16)W[k * N + n];
}

// ---------------- embedding MLP: 1 -> 64 -> 32 + LN (wave per row) ----------------
__global__ void emb_kernel(const float* __restrict__ cond,
                           const float* __restrict__ W1, const float* __restrict__ B1,
                           const float* __restrict__ W2, const float* __restrict__ B2,
                           float* __restrict__ emb, float* __restrict__ cons, int rows) {
    int row  = (blockIdx.x * blockDim.x + threadIdx.x) >> 5;
    int lane = threadIdx.x & 31;
    if (row >= rows) return;
    float c = cond[row];
    float acc = B2[lane];
#pragma unroll 8
    for (int j = 0; j < 64; ++j) {
        float h = fmaf(c, W1[j], B1[j]);
        h = h > 0.f ? h : 0.f;
        acc = fmaf(h, W2[j * FM + lane], acc);
    }
    float mu = acc;
    for (int o = 16; o; o >>= 1) mu += __shfl_xor(mu, o, 32);
    mu *= (1.f / 32.f);
    float d = acc - mu;
    float var = d * d;
    for (int o = 16; o; o >>= 1) var += __shfl_xor(var, o, 32);
    var *= (1.f / 32.f);
    float y = d * rsqrtf(var + EPS_C);
    emb[row * FM + lane]  = y;
    cons[row * FM + lane] = y;
}

// ---------------- edge scatter: dst[didx[e]] += src[sidx[e]] * ev[e] ----------------
// One edge per wave32; lane == feature (FM==32). Coalesced 128B row load,
// 32x global_atomic_add_f32. HBM/atomic-bound phase (~4.9GB/call total).
__global__ void scatter_kernel(const float* __restrict__ src, const float* __restrict__ ev,
                               const int* __restrict__ sidx, const int* __restrict__ didx,
                               float* __restrict__ dst, int n_edges) {
    int e    = blockIdx.x * (blockDim.x >> 5) + (threadIdx.x >> 5);
    int lane = threadIdx.x & 31;
    if (e >= n_edges) return;
    int s = sidx[e];
    int d = didx[e];
    float v = ev[e];
    atomicAdd(&dst[d * FM + lane], src[s * FM + lane] * v);
}

// ---------------- generic WMMA MLP: concat(NSEG x 32) -> 64 -> OUT (+LN or sigmoid) ----
// One 16-row tile per wave; 8 waves / 256-thread block. Weights arrive as
// pre-packed f16 fragments; activations staged through LDS (coalesced loads).
template <int NSEG, int OUT, bool DO_LN, bool DO_SIG>
__global__ __launch_bounds__(256)
void mlp_kernel(const float* __restrict__ x0, const float* __restrict__ x1,
                const float* __restrict__ x2,
                const _Float16* __restrict__ W1f, const float* __restrict__ B1p,
                const _Float16* __restrict__ W2f, const float* __restrict__ B2p,
                float* __restrict__ out, int rows) {
    constexpr int IN  = NSEG * 32;
    constexpr int HID = 64;
    constexpr int NT1 = HID / 16;   // hidden N-tiles
    constexpr int NT2 = OUT / 16;   // output N-tiles
    __shared__ _Float16 xs[8][16 * IN];    // staged f16 input rows
    __shared__ float    hs[8][16 * HID];   // hidden (then output) staging
    __shared__ float smu[8][16];
    __shared__ float sri[8][16];

    int wave = threadIdx.x >> 5;
    int lane = threadIdx.x & 31;
    int row0 = (blockIdx.x * 8 + wave) * 16;

    const float* segs[3] = { x0, x1, x2 };

    // ---- stage 16 rows of concat input into LDS (coalesced), convert to f16 ----
    for (int idx = lane; idx < 16 * IN; idx += 32) {
        int r = idx / IN, f = idx % IN;
        int grow = row0 + r;
        if (grow > rows - 1) grow = rows - 1;   // clamp; keep EXEC uniform
        const float* __restrict__ xp = segs[f >> 5];
        xs[wave][idx] = (_Float16)xp[grow * FM + (f & 31)];
    }
    __syncthreads();

    // ---- GEMM1: A(16 x IN) x W1(IN x 64), f16 in / f32 acc ----
    const v16h* __restrict__ w1v = (const v16h*)W1f;
    v8f acc[NT1];
#pragma unroll
    for (int t = 0; t < NT1; ++t) acc[t] = v8f{0,0,0,0,0,0,0,0};

#pragma unroll
    for (int c = 0; c < NSEG; ++c) {
        v16h af;
#pragma unroll
        for (int h = 0; h < 16; ++h)
            af[h] = xs[wave][(lane & 15) * IN + c * 32 + kmap(lane, h)];
#pragma unroll
        for (int t = 0; t < NT1; ++t) {
            v16h bf = w1v[(c * NT1 + t) * 32 + lane];   // contiguous 32B/lane
            acc[t] = __builtin_amdgcn_wmma_f32_16x16x32_f16(
                false, af, false, bf, (short)0, acc[t], false, false);
        }
    }

    // bias + relu -> LDS hidden tile (row-major 16 x 64)
#pragma unroll
    for (int t = 0; t < NT1; ++t) {
        int ncol = t * 16 + (lane & 15);
        float bv = B1p[ncol];
#pragma unroll
        for (int r = 0; r < 8; ++r) {
            int m = r + ((lane >> 4) << 3);
            float v = acc[t][r] + bv;
            hs[wave][m * HID + ncol] = v > 0.f ? v : 0.f;
        }
    }
    __syncthreads();

    // ---- GEMM2: hidden(16 x 64) x W2(64 x OUT) ----
    const v16h* __restrict__ w2v = (const v16h*)W2f;
    v8f acc2[NT2];
#pragma unroll
    for (int t = 0; t < NT2; ++t) acc2[t] = v8f{0,0,0,0,0,0,0,0};

#pragma unroll
    for (int c = 0; c < HID / 32; ++c) {
        v16h af;
#pragma unroll
        for (int h = 0; h < 16; ++h)
            af[h] = (_Float16)hs[wave][(lane & 15) * HID + c * 32 + kmap(lane, h)];
#pragma unroll
        for (int t = 0; t < NT2; ++t) {
            v16h bf = w2v[(c * NT2 + t) * 32 + lane];
            acc2[t] = __builtin_amdgcn_wmma_f32_16x16x32_f16(
                false, af, false, bf, (short)0, acc2[t], false, false);
        }
    }
    __syncthreads();

    // bias -> LDS output tile (16 x OUT), reusing hidden buffer
#pragma unroll
    for (int t = 0; t < NT2; ++t) {
        int ncol = t * 16 + (lane & 15);
        float bv = B2p[ncol];
#pragma unroll
        for (int r = 0; r < 8; ++r) {
            int m = r + ((lane >> 4) << 3);
            hs[wave][m * OUT + ncol] = acc2[t][r] + bv;
        }
    }
    __syncthreads();

    if (DO_LN) {
        if (lane < 16) {
            float mu = 0.f;
            for (int n = 0; n < OUT; ++n) mu += hs[wave][lane * OUT + n];
            mu /= (float)OUT;
            float var = 0.f;
            for (int n = 0; n < OUT; ++n) {
                float d = hs[wave][lane * OUT + n] - mu;
                var += d * d;
            }
            var /= (float)OUT;
            smu[wave][lane] = mu;
            sri[wave][lane] = rsqrtf(var + EPS_C);
        }
        __syncthreads();
        for (int idx = lane; idx < 16 * OUT; idx += 32) {
            int m = idx / OUT, n = idx % OUT;
            int grow = row0 + m;
            if (grow < rows)
                out[grow * OUT + n] = (hs[wave][m * OUT + n] - smu[wave][m]) * sri[wave][m];
        }
    } else {
        for (int idx = lane; idx < 16 * OUT; idx += 32) {
            int m = idx / OUT, n = idx % OUT;
            int grow = row0 + m;
            float v = hs[wave][m * OUT + n];
            if (DO_SIG) v = 1.f / (1.f + expf(-v));
            if (grow < rows) out[grow * OUT + n] = v;
        }
    }
}

// ---------------- host driver ----------------
extern "C" void kernel_launch(void* const* d_in, const int* in_sizes, int n_in,
                              void* d_out, int out_size, void* d_ws, size_t ws_size,
                              hipStream_t stream) {
    (void)in_sizes; (void)n_in; (void)out_size; (void)ws_size;

    const int*   edge_var   = (const int*)d_in[0];
    const int*   edge_const = (const int*)d_in[1];
    const float* edge_val   = (const float*)d_in[2];
    const float* cond       = (const float*)d_in[3];
    // d_in[4], d_in[5]: var_count/const_count scalars in device mem -> constants used
    const float* pc_W1 = (const float*)d_in[6];
    const float* pc_b1 = (const float*)d_in[7];
    const float* pc_W2 = (const float*)d_in[8];
    const float* pc_b2 = (const float*)d_in[9];
    const float* cu_W1 = (const float*)d_in[10];
    const float* cu_b1 = (const float*)d_in[11];
    const float* cu_W2 = (const float*)d_in[12];
    const float* cu_b2 = (const float*)d_in[13];
    const float* vu_W1 = (const float*)d_in[14];
    const float* vu_b1 = (const float*)d_in[15];
    const float* vu_W2 = (const float*)d_in[16];
    const float* vu_b2 = (const float*)d_in[17];
    const float* out_W1 = (const float*)d_in[18];
    const float* out_b1 = (const float*)d_in[19];
    const float* out_W2 = (const float*)d_in[20];
    const float* out_b2 = (const float*)d_in[21];
    float* out = (float*)d_out;

    // workspace carve-out (~45 MB + packed weights)
    char* ws = (char*)d_ws;
    size_t off = 0;
    auto carve = [&](size_t bytes) {
        void* p = (void*)(ws + off);
        off += (bytes + 255) & ~(size_t)255;
        return p;
    };
    float* variables   = (float*)carve((size_t)VAR_COUNT_C   * FM * sizeof(float));
    float* constraints = (float*)carve((size_t)CONST_COUNT_C * FM * sizeof(float));
    float* emb         = (float*)carve((size_t)CONST_COUNT_C * FM * sizeof(float));
    float* v2c         = (float*)carve((size_t)CONST_COUNT_C * FM * sizeof(float));
    float* c2v         = (float*)carve((size_t)VAR_COUNT_C   * FM * sizeof(float));
    _Float16* cu_W1f  = (_Float16*)carve((size_t)96 * 64 * sizeof(_Float16));
    _Float16* cu_W2f  = (_Float16*)carve((size_t)64 * 32 * sizeof(_Float16));
    _Float16* vu_W1f  = (_Float16*)carve((size_t)64 * 64 * sizeof(_Float16));
    _Float16* vu_W2f  = (_Float16*)carve((size_t)64 * 32 * sizeof(_Float16));
    _Float16* out_W1f = (_Float16*)carve((size_t)32 * 64 * sizeof(_Float16));
    _Float16* out_W2f = (_Float16*)carve((size_t)64 * 16 * sizeof(_Float16));

    // pack f32 weights into f16 WMMA fragment layout (tiny, once per launch)
    auto pack = [&](const float* W, _Float16* dst, int K, int N) {
        int total = (K >> 5) * (N >> 4) * 512;
        pack_wgt<<<(total + 255) / 256, 256, 0, stream>>>(W, dst, K, N);
    };
    pack(cu_W1,  cu_W1f,  96, 64);
    pack(cu_W2,  cu_W2f,  64, 32);
    pack(vu_W1,  vu_W1f,  64, 64);
    pack(vu_W2,  vu_W2f,  64, 32);
    pack(out_W1, out_W1f, 32, 64);
    pack(out_W2, out_W2f, 64, 16);

    // variables = ones
    {
        int n = VAR_COUNT_C * FM;
        fill_f32<<<(n + 255) / 256, 256, 0, stream>>>(variables, 1.0f, n);
    }
    // emb = MLP_LN(cond); constraints = emb
    emb_kernel<<<(CONST_COUNT_C + 7) / 8, 256, 0, stream>>>(
        cond, pc_W1, pc_b1, pc_W2, pc_b2, emb, constraints, CONST_COUNT_C);

    const int edge_blocks = (N_EDGES_C + 7) / 8;   // 8 edges (waves) per block
    for (int s = 0; s < PASS_STEPS; ++s) {
        hipMemsetAsync(v2c, 0, (size_t)CONST_COUNT_C * FM * sizeof(float), stream);
        scatter_kernel<<<edge_blocks, 256, 0, stream>>>(
            variables, edge_val, edge_var, edge_const, v2c, N_EDGES_C);
        mlp_kernel<3, 32, true, false><<<(CONST_COUNT_C + 127) / 128, 256, 0, stream>>>(
            constraints, emb, v2c, cu_W1f, cu_b1, cu_W2f, cu_b2, constraints, CONST_COUNT_C);

        hipMemsetAsync(c2v, 0, (size_t)VAR_COUNT_C * FM * sizeof(float), stream);
        scatter_kernel<<<edge_blocks, 256, 0, stream>>>(
            constraints, edge_val, edge_const, edge_var, c2v, N_EDGES_C);
        mlp_kernel<2, 32, true, false><<<(VAR_COUNT_C + 127) / 128, 256, 0, stream>>>(
            variables, c2v, nullptr, vu_W1f, vu_b1, vu_W2f, vu_b2, variables, VAR_COUNT_C);
    }

    // output head: 32 -> 64 -> 16, sigmoid
    mlp_kernel<1, 16, false, true><<<(VAR_COUNT_C + 127) / 128, 256, 0, stream>>>(
        variables, nullptr, nullptr, out_W1f, out_b1, out_W2f, out_b2, out, VAR_COUNT_C);
}
